// SelfAttention_1795296330010
// MI455X (gfx1250) — matrix operand order repaired and checked by reading, not compiled, not run on hardware
//
#include <hip/hip_runtime.h>

typedef __attribute__((ext_vector_type(16))) _Float16 v16h;
typedef __attribute__((ext_vector_type(8)))  _Float16 v8h;
typedef __attribute__((ext_vector_type(2)))  __fp16   v2fp;
typedef __attribute__((ext_vector_type(8)))  float    v8f;

#define NPIX 4096
#define CH   256
#define VSTR 40                       // padded LDS row stride (halves) = 80B, conflict-free
#define BUFH (CH * VSTR + 32 * VSTR)  // halves per LDS buffer: V panel + K panel

union V16 { v16h v; v8h h[2]; _Float16 e[16]; };
union PKU { v2fp v; unsigned u; };

// -------------------- Weight f32->f16 convert --------------------
__global__ __launch_bounds__(256) void sa_wconv_kernel(
    const float* __restrict__ Wq, const float* __restrict__ Wk,
    const float* __restrict__ Wv,
    _Float16* __restrict__ whq, _Float16* __restrict__ whk,
    _Float16* __restrict__ whv)
{
    const int i = blockIdx.x * 256 + threadIdx.x;   // 320 blocks -> 81920 elems
    if (i < 8192)        whq[i]         = (_Float16)Wq[i];
    else if (i < 16384)  whk[i - 8192]  = (_Float16)Wk[i - 8192];
    else                 whv[i - 16384] = (_Float16)Wv[i - 16384];
}

// -------------------- Transpose + convert: x[b][c][n] -> xh[b][n][c] f16 --------------------
__global__ __launch_bounds__(256) void sa_xpose_kernel(
    const float* __restrict__ x, _Float16* __restrict__ xh)
{
    __shared__ float xs[CH * 17];                   // stride 17 -> conflict-free
    const int b  = blockIdx.x >> 8;
    const int n0 = (blockIdx.x & 255) * 16;
    const int t  = threadIdx.x;

    const float* xp = x + ((size_t)(b * CH + t)) * NPIX + n0;
    #pragma unroll
    for (int j = 0; j < 16; ++j) xs[t * 17 + j] = xp[j];
    __syncthreads();

    const int j  = t >> 4;
    const int c0 = (t & 15) * 16;
    _Float16* op = xh + ((size_t)(b * NPIX + n0 + j)) * CH + c0;
    #pragma unroll
    for (int i = 0; i < 16; ++i) op[i] = (_Float16)xs[(c0 + i) * 17 + j];
}

// -------------------- WMMA projection kernel --------------------
// grid: B*32 blocks, 256 threads = 8 waves; wave w -> 16-pixel tile n = (bt*8+w)*16.
// q16/k16: [B][N][32] f16 (pixel-major rows); v16: [B][256][N] f16.
__global__ __launch_bounds__(256) void sa_wproj_kernel(
    const _Float16* __restrict__ xh,
    const _Float16* __restrict__ whq, const float* __restrict__ bq,
    const _Float16* __restrict__ whk, const float* __restrict__ bk,
    const _Float16* __restrict__ whv, const float* __restrict__ bv,
    _Float16* __restrict__ q16, _Float16* __restrict__ k16,
    _Float16* __restrict__ v16)
{
    const int b    = blockIdx.x >> 5;
    const int bt   = blockIdx.x & 31;
    const int wave = threadIdx.x >> 5;
    const int lane = threadIdx.x & 31;
    const int half = lane >> 4;
    const int nloc = lane & 15;
    const int n    = (bt * 8 + wave) * 16 + nloc;

    const v8f zf = {0.f, 0.f, 0.f, 0.f, 0.f, 0.f, 0.f, 0.f};

    // B operands: xh row, 8 k-blocks of 32 channels (held in 64 VGPRs)
    V16 xb[8];
    {
        const _Float16* xrow = xh + ((size_t)(b * NPIX + n)) * CH;
        #pragma unroll
        for (int kb = 0; kb < 8; ++kb) {
            xb[kb].h[0] = *(const v8h*)(xrow + kb * 32 + half * 16);
            xb[kb].h[1] = *(const v8h*)(xrow + kb * 32 + half * 16 + 8);
        }
    }

    // One 16x16 output tile: rows o0..o0+15 of W (row stride CH), full K=256
    auto tile = [&](const _Float16* wbase, int o0) -> v8f {
        v8f acc = zf;
        const _Float16* wr0 = wbase + ((size_t)(o0 + nloc)) * CH;
        #pragma unroll
        for (int kb = 0; kb < 8; ++kb) {
            V16 wa;   // A operand: row = o (lane&15), e[j]=W[o][kb*32+half*16+j]
            const _Float16* wr = wr0 + kb * 32;
            wa.h[0] = *(const v8h*)(wr + half * 16);
            wa.h[1] = *(const v8h*)(wr + half * 16 + 8);
            acc = __builtin_amdgcn_wmma_f32_16x16x32_f16(false, wa.v, false, xb[kb].v,
                                                         (short)0, acc, false, false);
        }
        return acc;
    };

    // Q and K: 2 o-tiles each; D layout row o = o0 + r + 8*half, col n = lane&15
    #pragma unroll
    for (int ot = 0; ot < 2; ++ot) {
        const v8f dq = tile(whq, ot * 16);
        const v8f dk = tile(whk, ot * 16);
        _Float16* qp = q16 + ((size_t)(b * NPIX + n)) * 32;
        _Float16* kp = k16 + ((size_t)(b * NPIX + n)) * 32;
        #pragma unroll
        for (int r = 0; r < 8; ++r) {
            const int o = ot * 16 + r + 8 * half;
            qp[o] = (_Float16)(dq[r] + bq[o]);
            kp[o] = (_Float16)(dk[r] + bk[o]);
        }
    }

    // V: 16 o-tiles
    #pragma unroll
    for (int ot = 0; ot < 16; ++ot) {
        const v8f dv = tile(whv, ot * 16);
        #pragma unroll
        for (int r = 0; r < 8; ++r) {
            const int o = ot * 16 + r + 8 * half;
            v16[((size_t)(b * CH + o)) * NPIX + n] = (_Float16)(dv[r] + bv[o]);
        }
    }
}

// Async DMA: 16 bytes global -> LDS (ASYNCcnt-tracked). lds_off = workgroup LDS byte address.
__device__ __forceinline__ void async_copy16(unsigned lds_off, const void* gaddr) {
    asm volatile("global_load_async_to_lds_b128 %0, %1, off"
                 :: "v"(lds_off), "v"((unsigned long long)gaddr)
                 : "memory");
}

// -------------------- Fused flash-attention kernel --------------------
// grid: B * (N/128) blocks, 256 threads = 8 waves; wave w owns query tile (bt*8+w)*16.
// Per 32-key block: K (2KB) + V (16KB) staged once into LDS (async, double-buffered),
// then each wave: St = K_rows x Q^T (2 WMMA), online softmax, O^T += V x P^T (16 WMMA).
__global__ __launch_bounds__(256) void sa_attn_kernel(
    const _Float16* __restrict__ q16, const _Float16* __restrict__ k16,
    const _Float16* __restrict__ v16, const float* __restrict__ x,
    const float* __restrict__ gamma, float* __restrict__ out)
{
    __shared__ _Float16 smem[2 * BUFH];

    const int b    = blockIdx.x >> 5;          // 32 blocks per batch
    const int bt   = blockIdx.x & 31;
    const int wave = threadIdx.x >> 5;
    const int lane = threadIdx.x & 31;
    const int half = lane >> 4;
    const int mloc = lane & 15;
    const int m    = (bt * 8 + wave) * 16 + mloc;
    const int t    = threadIdx.x;

    // Stage one 32-key block (V: 256 rows x 32 halves, K: 32 rows x 32 halves) into buf.
    auto stage = [&](int buf, int n0) {
        const unsigned base = (unsigned)(uintptr_t)&smem[buf * BUFH];
        const _Float16* vg = v16 + ((size_t)(b * CH + t)) * NPIX + n0;
        const unsigned  vl = base + (unsigned)(t * VSTR) * 2;
        #pragma unroll
        for (int ch = 0; ch < 4; ++ch)
            async_copy16(vl + ch * 16, vg + ch * 8);
        if (t < 128) {
            const int row = t >> 2, ch = t & 3;
            const _Float16* kg = k16 + ((size_t)(b * NPIX + n0 + row)) * 32 + ch * 8;
            async_copy16(base + (unsigned)((CH + row) * VSTR + ch * 8) * 2, kg);
        }
    };

    // Q tile as B operand: e[j] = Q[m][half*16 + j] (contiguous 16 halves)
    V16 qb;
    {
        const _Float16* qrow = q16 + ((size_t)(b * NPIX + m)) * 32 + half * 16;
        qb.h[0] = *(const v8h*)(qrow);
        qb.h[1] = *(const v8h*)(qrow + 8);
    }

    const v8f zf = {0.f, 0.f, 0.f, 0.f, 0.f, 0.f, 0.f, 0.f};
    v8f ot[16];
    #pragma unroll
    for (int ct = 0; ct < 16; ++ct) ot[ct] = zf;

    float rm = -3.0e38f;   // running row max
    float ls = 0.0f;       // running row sum

    stage(0, 0);

    for (int n0 = 0; n0 < NPIX; n0 += 32) {
        const int buf = (n0 >> 5) & 1;
        asm volatile("s_wait_asynccnt 0x0" ::: "memory");  // my DMA done
        __syncthreads();                                   // everyone's DMA done
        if (n0 + 32 < NPIX) stage(buf ^ 1, n0 + 32);       // overlap next block's DMA

        const _Float16* vbase = smem + buf * BUFH;
        const _Float16* kbase = vbase + CH * VSTR;

        // K tiles as A operands: row = key (lane&15), e[j]=K[n][half*16+j]
        V16 ka0, ka1;
        {
            const _Float16* kr0 = kbase + mloc * VSTR;
            const _Float16* kr1 = kbase + (16 + mloc) * VSTR;
            ka0.h[0] = *(const v8h*)(kr0 + half * 16);
            ka0.h[1] = *(const v8h*)(kr0 + half * 16 + 8);
            ka1.h[0] = *(const v8h*)(kr1 + half * 16);
            ka1.h[1] = *(const v8h*)(kr1 + half * 16 + 8);
        }

        v8f st0 = __builtin_amdgcn_wmma_f32_16x16x32_f16(false, ka0.v, false, qb.v,
                                                         (short)0, zf, false, false);
        v8f st1 = __builtin_amdgcn_wmma_f32_16x16x32_f16(false, ka1.v, false, qb.v,
                                                         (short)0, zf, false, false);
        // Lane holds, for query m: st0[r] -> key n0+half*8+r, st1[r] -> n0+16+half*8+r.

        float bm = -3.0e38f;
        #pragma unroll
        for (int r = 0; r < 8; ++r) { bm = fmaxf(bm, st0[r]); bm = fmaxf(bm, st1[r]); }
        bm = fmaxf(bm, __shfl_xor(bm, 16, 32));
        const float nm   = fmaxf(rm, bm);
        const float corr = __expf(rm - nm);
        rm = nm;

        float psum = 0.0f;
        unsigned pk0[4], pk1[4];
        #pragma unroll
        for (int i = 0; i < 4; ++i) {
            const float a0 = __expf(st0[2 * i]     - nm);
            const float a1 = __expf(st0[2 * i + 1] - nm);
            const float b0 = __expf(st1[2 * i]     - nm);
            const float b1 = __expf(st1[2 * i + 1] - nm);
            psum += (a0 + a1) + (b0 + b1);
            PKU u0; u0.v = __builtin_amdgcn_cvt_pkrtz(a0, a1); pk0[i] = u0.u;
            PKU u1; u1.v = __builtin_amdgcn_cvt_pkrtz(b0, b1); pk1[i] = u1.u;
        }
        psum += __shfl_xor(psum, 16, 32);
        ls = ls * corr + psum;

        // P^T as B operand: e[j] = P[m][n0 + half*16 + j]; partner lane holds the
        // complementary 16 keys -> exchange packed f16 pairs (8 dword shuffles).
        V16 pb;
        #pragma unroll
        for (int i = 0; i < 4; ++i) {
            const unsigned s0 = __shfl_xor(pk0[i], 16, 32);
            const unsigned s1 = __shfl_xor(pk1[i], 16, 32);
            ((unsigned*)&pb)[i]     = half ? s1 : pk0[i];   // keys base+ 0..7  (this half)
            ((unsigned*)&pb)[4 + i] = half ? pk1[i] : s0;   // keys base+ 8..15 (this half)
        }

        // Rescale accumulators by correction factor
        #pragma unroll
        for (int ct = 0; ct < 16; ++ct) ot[ct] = ot[ct] * corr;

        // O^T accumulation: 16 c-tiles x (16x16x32) WMMA, V from LDS
        #pragma unroll
        for (int ct = 0; ct < 16; ++ct) {
            V16 va;   // A operand: row = channel (lane&15) within tile
            const _Float16* vr = vbase + (ct * 16 + mloc) * VSTR;
            va.h[0] = *(const v8h*)(vr + half * 16);
            va.h[1] = *(const v8h*)(vr + half * 16 + 8);
            ot[ct] = __builtin_amdgcn_wmma_f32_16x16x32_f16(false, va.v, false, pb.v,
                                                            (short)0, ot[ct], false, false);
        }
    }

    // Epilogue: out = gamma * O / ls + x  (C/D layout: c row = ct*16 + r + 8*half)
    const float inv = 1.0f / ls;
    const float g   = gamma[0];
    #pragma unroll
    for (int ct = 0; ct < 16; ++ct) {
        #pragma unroll
        for (int r = 0; r < 8; ++r) {
            const int c = ct * 16 + r + 8 * half;
            const size_t idx = ((size_t)(b * CH + c)) * NPIX + m;
            out[idx] = fmaf(g, ot[ct][r] * inv, x[idx]);
        }
    }
}

extern "C" void kernel_launch(void* const* d_in, const int* in_sizes, int n_in,
                              void* d_out, int out_size, void* d_ws, size_t ws_size,
                              hipStream_t stream) {
    const float* x     = (const float*)d_in[0];
    const float* Wq    = (const float*)d_in[1];
    const float* bq    = (const float*)d_in[2];
    const float* Wk    = (const float*)d_in[3];
    const float* bk    = (const float*)d_in[4];
    const float* Wv    = (const float*)d_in[5];
    const float* bv    = (const float*)d_in[6];
    const float* gamma = (const float*)d_in[7];
    float* out = (float*)d_out;

    _Float16* q16 = (_Float16*)d_ws;                         // 8*4096*32  = 2 MB
    _Float16* k16 = q16 + (size_t)8 * NPIX * 32;             // 2 MB
    _Float16* v16 = k16 + (size_t)8 * NPIX * 32;             // 16 MB
    _Float16* xh  = v16 + (size_t)8 * CH * NPIX;             // 16 MB
    _Float16* whq = xh  + (size_t)8 * NPIX * CH;             // 16 KB
    _Float16* whk = whq + (size_t)32 * CH;                   // 16 KB
    _Float16* whv = whk + (size_t)32 * CH;                   // 128 KB

    sa_wconv_kernel<<<320, 256, 0, stream>>>(Wq, Wk, Wv, whq, whk, whv);
    sa_xpose_kernel<<<8 * (NPIX / 16), 256, 0, stream>>>(x, xh);
    sa_wproj_kernel<<<8 * (NPIX / 128), 256, 0, stream>>>(
        xh, whq, bq, whk, bk, whv, bv, q16, k16, v16);
    sa_attn_kernel<<<8 * (NPIX / 128), 256, 0, stream>>>(
        q16, k16, v16, x, gamma, out);
}
